// StockGRUModel_67190468379297
// MI455X (gfx1250) — compile-verified, matrix operand order checked by reading, not run on hardware
//
#include <hip/hip_runtime.h>
#include <hip/hip_bf16.h>
#include <math.h>

// ---------------------------------------------------------------------------
// StockGRUModel on MI455X (gfx1250, wave32, WMMA).
// B=256, T=512, F=32, H=512, E=32, OUT=30.
//
// gru2_wmma: persistent kernel, grid = 8 row-groups x 16 col-WGs = 128 blocks,
// 128 threads (4 waves). Wave owns a 16x16 tile of h (fp32 in registers for
// both layers, carried across all 512 steps). Per-WG weight slices (294 KB)
// live in LDS for all 512 timesteps; bf16 copies of h are exchanged through
// L2 (double buffered) with a 16-member atomic barrier per column group,
// twice per timestep. All GEMMs: v_wmma_f32_16x16x32_bf16.
// ---------------------------------------------------------------------------

typedef __attribute__((ext_vector_type(16))) __bf16       v16bf;
typedef __attribute__((ext_vector_type(8)))  float        v8f;
typedef __attribute__((ext_vector_type(4)))  unsigned int uix4;

union FragU { v16bf v; uix4 q[2]; };

// A-fragment (16x32 bf16) / B-fragment (32x16 bf16, one W row per column):
// lane<16 : row = row0+lane,    K = k0 + {0..7} and k0 + {16..23}
// lane>=16: row = row0+lane-16, K = k0 + {8..15} and k0 + {24..31}
__device__ __forceinline__ v16bf load_frag(const __bf16* __restrict__ base,
                                           int row0, int rowlen, int k0, int lane) {
  const int r = row0 + (lane & 15);
  const int k = k0 + ((lane >> 4) << 3);
  const __bf16* a0 = base + (size_t)r * (size_t)rowlen + k;
  FragU f;
  f.q[0] = *reinterpret_cast<const uix4*>(a0);
  f.q[1] = *reinterpret_cast<const uix4*>(a0 + 16);
  return f.v;
}

__device__ __forceinline__ v8f wmma_bf16(v16bf a, v16bf b, v8f c) {
  return __builtin_amdgcn_wmma_f32_16x16x32_bf16(false, a, false, b, (short)0, c,
                                                 false, false);
}

__device__ __forceinline__ v8f splat8(float v) {
  v8f r;
#pragma unroll
  for (int i = 0; i < 8; ++i) r[i] = v;
  return r;
}

// Fast gate nonlinearities: v_exp_f32 + v_rcp_f32; saturate correctly at +-inf.
__device__ __forceinline__ float sigmoidf_(float x) {
  return __builtin_amdgcn_rcpf(1.0f + __expf(-x));
}
__device__ __forceinline__ float tanhf_(float x) {
  return 1.0f - 2.0f * __builtin_amdgcn_rcpf(1.0f + __expf(2.0f * x));
}

// 16-workgroup sense-reversing barrier over L2 atomics (per column group).
__device__ __forceinline__ void group_barrier(unsigned* cnt, unsigned* gen,
                                              unsigned members) {
  __builtin_amdgcn_fence(__ATOMIC_RELEASE, "agent");  // flush this wave's stores
  __syncthreads();
  if (threadIdx.x == 0) {
    unsigned g = __hip_atomic_load(gen, __ATOMIC_RELAXED, __HIP_MEMORY_SCOPE_AGENT);
    unsigned a = __hip_atomic_fetch_add(cnt, 1u, __ATOMIC_ACQ_REL,
                                        __HIP_MEMORY_SCOPE_AGENT);
    if (a + 1u == members) {
      __hip_atomic_store(cnt, 0u, __ATOMIC_RELAXED, __HIP_MEMORY_SCOPE_AGENT);
      __hip_atomic_fetch_add(gen, 1u, __ATOMIC_RELEASE, __HIP_MEMORY_SCOPE_AGENT);
    } else {
      while (__hip_atomic_load(gen, __ATOMIC_ACQUIRE,
                               __HIP_MEMORY_SCOPE_AGENT) == g) {
        __builtin_amdgcn_s_sleep(2);
      }
    }
  }
  __syncthreads();
  __builtin_amdgcn_fence(__ATOMIC_ACQUIRE, "agent");  // invalidate L0 for h reads
}

// ---------------------------------------------------------------------------
__global__ void f32_to_bf16(const float* __restrict__ src, __bf16* __restrict__ dst,
                            int n) {
  int i = blockIdx.x * blockDim.x + threadIdx.x;
  if (i < n) dst[i] = (__bf16)src[i];
}

// Gate packing [r,z,n]: rz combined (bih+bhh), n split (x-part bias / h-part bias)
__global__ void bias_prep(const float* __restrict__ bih, const float* __restrict__ bhh,
                          float* __restrict__ rz, float* __restrict__ xn,
                          float* __restrict__ hn) {
  int c = blockIdx.x * blockDim.x + threadIdx.x;
  if (c < 512) {
    rz[c]       = bih[c]        + bhh[c];
    rz[512 + c] = bih[512 + c]  + bhh[512 + c];
    xn[c]       = bih[1024 + c];
    hn[c]       = bhh[1024 + c];
  }
}

// ---------------------------------------------------------------------------
// LDS weight cache layout (per WG, bf16 elements):
//   [0      , 49152) : Whh0 slice, 96 rows (3 gates x 32 cols) x 512 K
//   [49152  , 98304) : Wih1 slice
//   [98304  ,147456) : Whh1 slice
//   [147456 ,150528) : Wih0 slice, 96 rows x 32 K
#define SW_WHH0 0
#define SW_WIH1 49152
#define SW_WHH1 98304
#define SW_WIH0 147456
#define SW_TOTAL 150528

__global__ __launch_bounds__(128) void gru2_wmma(
    const __bf16* __restrict__ xbf,                       // (B,T,F) bf16
    const __bf16* __restrict__ wih0,                      // (1536,32)
    const __bf16* __restrict__ whh0,                      // (1536,512)
    const __bf16* __restrict__ wih1,                      // (1536,512)
    const __bf16* __restrict__ whh1,                      // (1536,512)
    const float* __restrict__ b0rz, const float* __restrict__ b0xn,
    const float* __restrict__ b0hn,
    const float* __restrict__ b1rz, const float* __restrict__ b1xn,
    const float* __restrict__ b1hn,
    __bf16* __restrict__ h0g,                             // 2 x (256,512) bf16
    __bf16* __restrict__ h1g,                             // 2 x (256,512) bf16
    float* __restrict__ h1f,                              // (256,512) f32 out
    unsigned* __restrict__ syncw) {
  __shared__ __align__(16) __bf16 smem[SW_TOTAL];

  const int lane = threadIdx.x & 31;
  const int wid  = threadIdx.x >> 5;
  const int grp  = blockIdx.x >> 4;   // 0..7  : 32 batch rows
  const int cwg  = blockIdx.x & 15;   // 0..15 : 32 h columns
  const int rbase = grp * 32 + (wid >> 1) * 16;
  const int cbase = cwg * 32 + (wid & 1) * 16;
  const int lcb   = (wid & 1) * 16;    // local column base inside WG slice
  const int col   = cbase + (lane & 15);
  const int rofs  = (lane >> 4) << 3;  // +8 rows for upper half-wave in C layout

  // ---- stage per-WG weight slices into LDS (once; reused 512 steps) ----
  {
    const __bf16* gsrc[3] = {whh0, wih1, whh1};
    __bf16* gdst[3] = {smem + SW_WHH0, smem + SW_WIH1, smem + SW_WHH1};
    for (int m = 0; m < 3; ++m) {
      const __bf16* src = gsrc[m];
      __bf16* dst = gdst[m];
      for (int idx = threadIdx.x; idx < 96 * 64; idx += 128) {
        const int row  = idx >> 6;                              // 0..95
        const int ch   = idx & 63;                              // 16B chunk
        const int grow = (row >> 5) * 512 + cwg * 32 + (row & 31);
        *reinterpret_cast<uix4*>(dst + row * 512 + ch * 8) =
            *reinterpret_cast<const uix4*>(src + (size_t)grow * 512 + ch * 8);
      }
    }
    for (int idx = threadIdx.x; idx < 96 * 4; idx += 128) {
      const int row  = idx >> 2;
      const int ch   = idx & 3;
      const int grow = (row >> 5) * 512 + cwg * 32 + (row & 31);
      *reinterpret_cast<uix4*>(smem + SW_WIH0 + row * 32 + ch * 8) =
          *reinterpret_cast<const uix4*>(wih0 + (size_t)grow * 32 + ch * 8);
    }
  }
  __syncthreads();

  const __bf16* s_whh0 = smem + SW_WHH0;
  const __bf16* s_wih1 = smem + SW_WIH1;
  const __bf16* s_whh1 = smem + SW_WHH1;
  const __bf16* s_wih0 = smem + SW_WIH0;

  unsigned* cnt = syncw + grp * 32;          // one cacheline per group
  unsigned* gen = syncw + 256 + grp * 32;

  const float c_b0r = b0rz[col], c_b0z = b0rz[512 + col];
  const float c_b0x = b0xn[col], c_b0h = b0hn[col];
  const float c_b1r = b1rz[col], c_b1z = b1rz[512 + col];
  const float c_b1x = b1xn[col], c_b1h = b1hn[col];

  v8f h0 = splat8(0.0f);
  v8f h1 = splat8(0.0f);

  // software-pipelined input-projection A fragment (for step t)
  v16bf ax = load_frag(xbf, rbase, 512 * 32, 0, lane);

  for (int t = 0; t < 512; ++t) {
    const __bf16* h0rbuf = h0g + (size_t)(t & 1) * (256 * 512);
    __bf16*       h0wbuf = h0g + (size_t)((t + 1) & 1) * (256 * 512);
    const __bf16* h1rbuf = h1g + (size_t)(t & 1) * (256 * 512);
    __bf16*       h1wbuf = h1g + (size_t)((t + 1) & 1) * (256 * 512);

    // ---------------- layer 0 ----------------
    v8f ar = splat8(c_b0r), az = splat8(c_b0z);
    v8f axn = splat8(c_b0x), ahn = splat8(c_b0h);
    {  // input projection, K = F = 32 in one WMMA step
      ar  = wmma_bf16(ax, load_frag(s_wih0,      lcb, 32, 0, lane), ar);
      az  = wmma_bf16(ax, load_frag(s_wih0, 32 + lcb, 32, 0, lane), az);
      axn = wmma_bf16(ax, load_frag(s_wih0, 64 + lcb, 32, 0, lane), axn);
    }
#pragma unroll 4
    for (int kk = 0; kk < 512; kk += 32) {  // recurrent projection
      v16bf ah = load_frag(h0rbuf, rbase, 512, kk, lane);
      ar  = wmma_bf16(ah, load_frag(s_whh0,      lcb, 512, kk, lane), ar);
      az  = wmma_bf16(ah, load_frag(s_whh0, 32 + lcb, 512, kk, lane), az);
      ahn = wmma_bf16(ah, load_frag(s_whh0, 64 + lcb, 512, kk, lane), ahn);
    }
#pragma unroll
    for (int i = 0; i < 8; ++i) {
      float r = sigmoidf_(ar[i]);
      float z = sigmoidf_(az[i]);
      float n = tanhf_(axn[i] + r * ahn[i]);
      h0[i] = (1.0f - z) * n + z * h0[i];
    }
#pragma unroll
    for (int i = 0; i < 8; ++i) {
      int row = rbase + i + rofs;
      h0wbuf[row * 512 + col] = (__bf16)h0[i];
    }
    group_barrier(cnt, gen, 16u);

    // ---------------- layer 1 (merged ih/hh K-loop: 6 WMMA streams) --------
    ar = splat8(c_b1r); az = splat8(c_b1z);
    axn = splat8(c_b1x); ahn = splat8(c_b1h);
#pragma unroll 2
    for (int kk = 0; kk < 512; kk += 32) {
      v16bf a0 = load_frag(h0wbuf, rbase, 512, kk, lane);  // fresh h0
      v16bf a1 = load_frag(h1rbuf, rbase, 512, kk, lane);  // previous h1
      ar  = wmma_bf16(a0, load_frag(s_wih1,      lcb, 512, kk, lane), ar);
      az  = wmma_bf16(a0, load_frag(s_wih1, 32 + lcb, 512, kk, lane), az);
      axn = wmma_bf16(a0, load_frag(s_wih1, 64 + lcb, 512, kk, lane), axn);
      ar  = wmma_bf16(a1, load_frag(s_whh1,      lcb, 512, kk, lane), ar);
      az  = wmma_bf16(a1, load_frag(s_whh1, 32 + lcb, 512, kk, lane), az);
      ahn = wmma_bf16(a1, load_frag(s_whh1, 64 + lcb, 512, kk, lane), ahn);
    }
#pragma unroll
    for (int i = 0; i < 8; ++i) {
      float r = sigmoidf_(ar[i]);
      float z = sigmoidf_(az[i]);
      float n = tanhf_(axn[i] + r * ahn[i]);
      h1[i] = (1.0f - z) * n + z * h1[i];
    }
#pragma unroll
    for (int i = 0; i < 8; ++i) {
      int row = rbase + i + rofs;
      h1wbuf[row * 512 + col] = (__bf16)h1[i];
    }
    // prefetch next step's x fragment; in flight across the barrier wait
    ax = load_frag(xbf + (size_t)((t + 1) & 511) * 32, rbase, 512 * 32, 0, lane);
    group_barrier(cnt, gen, 16u);
  }

  // final top-layer hidden state in fp32 for the FC head
#pragma unroll
  for (int i = 0; i < 8; ++i) {
    int row = rbase + i + rofs;
    h1f[row * 512 + col] = h1[i];
  }
}

// ---------------------------------------------------------------------------
// FC head: concat(h_last, embed) -> relu(fc1) -> fc2. Tiny (~0.15 GFLOP).
__global__ __launch_bounds__(256) void head_fc(
    const float* __restrict__ h1f, const int* __restrict__ ticker,
    const float* __restrict__ emb, const float* __restrict__ fc1w,
    const float* __restrict__ fc1b, const float* __restrict__ fc2w,
    const float* __restrict__ fc2b, float* __restrict__ out) {
  __shared__ float sin_[544];
  __shared__ float sh_[512];
  const int b = blockIdx.x;
  for (int i = threadIdx.x; i < 512; i += 256) sin_[i] = h1f[b * 512 + i];
  if (threadIdx.x < 32) sin_[512 + threadIdx.x] = emb[ticker[b] * 32 + threadIdx.x];
  __syncthreads();
  for (int j = threadIdx.x; j < 512; j += 256) {
    float acc = fc1b[j];
    const float* w = fc1w + (size_t)j * 544;
#pragma unroll 8
    for (int k = 0; k < 544; ++k) acc = fmaf(sin_[k], w[k], acc);
    sh_[j] = fmaxf(acc, 0.0f);
  }
  __syncthreads();
  for (int j = threadIdx.x; j < 30; j += 256) {
    float acc = fc2b[j];
    const float* w = fc2w + (size_t)j * 512;
#pragma unroll 8
    for (int k = 0; k < 512; ++k) acc = fmaf(sh_[k], w[k], acc);
    out[b * 30 + j] = acc;
  }
}

// ---------------------------------------------------------------------------
extern "C" void kernel_launch(void* const* d_in, const int* in_sizes, int n_in,
                              void* d_out, int out_size, void* d_ws, size_t ws_size,
                              hipStream_t stream) {
  (void)in_sizes; (void)n_in; (void)out_size; (void)ws_size;
  const int*   ticker = (const int*)d_in[0];
  const float* seq    = (const float*)d_in[1];
  const float* emb    = (const float*)d_in[2];
  const float* Wih0   = (const float*)d_in[3];
  const float* Whh0   = (const float*)d_in[4];
  const float* bih0   = (const float*)d_in[5];
  const float* bhh0   = (const float*)d_in[6];
  const float* Wih1   = (const float*)d_in[7];
  const float* Whh1   = (const float*)d_in[8];
  const float* bih1   = (const float*)d_in[9];
  const float* bhh1   = (const float*)d_in[10];
  const float* fc1w   = (const float*)d_in[11];
  const float* fc1b   = (const float*)d_in[12];
  const float* fc2w   = (const float*)d_in[13];
  const float* fc2b   = (const float*)d_in[14];

  char* ws = (char*)d_ws;
  size_t cur = 0;
  auto alloc = [&](size_t bytes) {
    size_t o = cur;
    cur = (cur + bytes + 255) & ~(size_t)255;
    return o;
  };
  const size_t o_sync = alloc(2048);                 // barrier cnt/gen, 8 groups
  const size_t o_h0   = alloc(2 * 256 * 512 * 2);    // double-buffered bf16 h0
  const size_t o_h1   = alloc(2 * 256 * 512 * 2);    // double-buffered bf16 h1
  const size_t zeroBytes = cur;                      // zero sync + h buffers
  const size_t o_wih0 = alloc(1536 * 32 * 2);
  const size_t o_whh0 = alloc(1536 * 512 * 2);
  const size_t o_wih1 = alloc(1536 * 512 * 2);
  const size_t o_whh1 = alloc(1536 * 512 * 2);
  const size_t o_x    = alloc((size_t)256 * 512 * 32 * 2);
  const size_t o_b0   = alloc(2048 * 4);             // rz[1024], xn[512], hn[512]
  const size_t o_b1   = alloc(2048 * 4);
  const size_t o_h1f  = alloc(256 * 512 * 4);

  hipMemsetAsync(ws, 0, zeroBytes, stream);

  f32_to_bf16<<<(1536 * 32 + 255) / 256, 256, 0, stream>>>(
      Wih0, (__bf16*)(ws + o_wih0), 1536 * 32);
  f32_to_bf16<<<(1536 * 512 + 255) / 256, 256, 0, stream>>>(
      Whh0, (__bf16*)(ws + o_whh0), 1536 * 512);
  f32_to_bf16<<<(1536 * 512 + 255) / 256, 256, 0, stream>>>(
      Wih1, (__bf16*)(ws + o_wih1), 1536 * 512);
  f32_to_bf16<<<(1536 * 512 + 255) / 256, 256, 0, stream>>>(
      Whh1, (__bf16*)(ws + o_whh1), 1536 * 512);
  f32_to_bf16<<<(256 * 512 * 32 + 255) / 256, 256, 0, stream>>>(
      seq, (__bf16*)(ws + o_x), 256 * 512 * 32);

  float* b0 = (float*)(ws + o_b0);
  float* b1 = (float*)(ws + o_b1);
  bias_prep<<<2, 256, 0, stream>>>(bih0, bhh0, b0, b0 + 1024, b0 + 1536);
  bias_prep<<<2, 256, 0, stream>>>(bih1, bhh1, b1, b1 + 1024, b1 + 1536);

  gru2_wmma<<<128, 128, 0, stream>>>(
      (const __bf16*)(ws + o_x),
      (const __bf16*)(ws + o_wih0), (const __bf16*)(ws + o_whh0),
      (const __bf16*)(ws + o_wih1), (const __bf16*)(ws + o_whh1),
      b0, b0 + 1024, b0 + 1536, b1, b1 + 1024, b1 + 1536,
      (__bf16*)(ws + o_h0), (__bf16*)(ws + o_h1),
      (float*)(ws + o_h1f), (unsigned*)(ws + o_sync));

  head_fc<<<256, 256, 0, stream>>>((const float*)(ws + o_h1f), ticker, emb,
                                   fc1w, fc1b, fc2w, fc2b, (float*)d_out);
}
